// LocalisationNet2_65146063946564
// MI455X (gfx1250) — matrix operand order repaired and checked by reading, not compiled, not run on hardware
//
#include <hip/hip_runtime.h>
#include <hip/hip_bf16.h>

typedef __bf16 bf16_t;
typedef __attribute__((ext_vector_type(16))) __bf16 v16bf;
typedef __attribute__((ext_vector_type(8)))  __bf16 v8bf;
typedef __attribute__((ext_vector_type(8)))  float  v8f;
typedef __attribute__((ext_vector_type(8)))  unsigned int v8u;

#define EPS 1e-5f
#define Bn 64
#define H0 224
#define W0 224
#define C1 24
#define H1 112
#define W1 112
#define C2 48
#define H2 56
#define W2 56
#define NCHUNK 7              /* K2 = 216 padded to 224 = 7*32 */
#define KF (C2*H2*W2)         /* 150528 */

/* small fp32 region of workspace (float indices) */
#define OFF_SUM1  0
#define OFF_SQ1   24
#define OFF_MEAN1 48
#define OFF_RS1   72
#define OFF_SUM2  96
#define OFF_SQ2   144
#define OFF_MEAN2 192
#define OFF_RS2   240
#define OFF_FEAT  288         /* 64*48 = 3072 */
#define OFF_THETA 3360        /* 64*6 */
#define ZERO_COUNT 3360

/* workspace byte offsets (all 16B-aligned) */
#define WPACK_BYTE   16384ull                                   /* 3*7*32*16 bf16 = 21504 B */
#define POOLED1_BYTE 65536ull                                   /* 64*112*112*24 bf16 = 38.5 MB */
#define CONV2RAW_BYTE (POOLED1_BYTE + (size_t)Bn*H1*W1*C1*2)    /* 64*48*112*112 bf16 = 77 MB */
#define X1_BYTE       (CONV2RAW_BYTE + (size_t)Bn*C2*H1*W1*2)   /* 64*150528 bf16 = 19.3 MB */

__device__ __forceinline__ unsigned int f32_to_bf16_rne(float f) {
  unsigned int u = __float_as_uint(f);
  return (u + 0x7fffu + ((u >> 16) & 1u)) >> 16;
}

__global__ void k_zero(float* __restrict__ p, int n) {
  int i = blockIdx.x * blockDim.x + threadIdx.x;
  if (i < n) p[i] = 0.f;
}

/* ---- pass 1: conv1 recomputed on the fly, accumulate per-channel sum/sumsq ---- */
__global__ void k_conv1_stats(const float* __restrict__ x, const float* __restrict__ w,
                              const float* __restrict__ bias, float* __restrict__ F) {
  __shared__ float sw[C1 * 9], sb[C1], ssum[C1], ssq[C1];
  int t = threadIdx.x;
  if (t < C1 * 9) sw[t] = w[t];
  if (t < C1) { sb[t] = bias[t]; ssum[t] = 0.f; ssq[t] = 0.f; }
  __syncthreads();
  int gid = blockIdx.x * 256 + t;
  int b = gid / (H0 * W0); int r = gid % (H0 * W0);
  int y = r / W0; int xx = r % W0;
  const float* xb = x + (size_t)b * H0 * W0;
  float tap[9];
#pragma unroll
  for (int kh = 0; kh < 3; kh++)
#pragma unroll
    for (int kw = 0; kw < 3; kw++) {
      int yy = y + kh - 1, xc = xx + kw - 1;
      tap[kh * 3 + kw] = (yy >= 0 && yy < H0 && xc >= 0 && xc < W0) ? xb[yy * W0 + xc] : 0.f;
    }
  for (int c = 0; c < C1; c++) {
    float a = sb[c];
#pragma unroll
    for (int i = 0; i < 9; i++) a = fmaf(sw[c * 9 + i], tap[i], a);
    atomicAdd(&ssum[c], a);
    atomicAdd(&ssq[c], a * a);
  }
  __syncthreads();
  if (t < C1) { atomicAdd(&F[OFF_SUM1 + t], ssum[t]); atomicAdd(&F[OFF_SQ1 + t], ssq[t]); }
}

__global__ void k_finalize(float* __restrict__ F, int sumoff, int sqoff, int meanoff,
                           int rsoff, int nchan, float inv_n) {
  int t = threadIdx.x;
  if (t < nchan) {
    float m = F[sumoff + t] * inv_n;
    float v = F[sqoff + t] * inv_n - m * m;
    F[meanoff + t] = m;
    F[rsoff + t]  = rsqrtf(v + EPS);
  }
}

/* ---- pass 2: conv1 + BN + ReLU + maxpool2, write pooled1 as NHWC bf16 ---- */
__global__ void k_conv1_apply(const float* __restrict__ x, const float* __restrict__ w,
                              const float* __restrict__ bias, const float* __restrict__ g,
                              const float* __restrict__ beta, const float* __restrict__ F,
                              bf16_t* __restrict__ pooled1) {
  __shared__ float sw[C1 * 9], sb[C1], sm[C1], sr[C1], sg[C1], sbe[C1];
  int t = threadIdx.x;
  if (t < C1 * 9) sw[t] = w[t];
  if (t < C1) {
    sb[t] = bias[t]; sm[t] = F[OFF_MEAN1 + t]; sr[t] = F[OFF_RS1 + t];
    sg[t] = g[t];    sbe[t] = beta[t];
  }
  __syncthreads();
  int gid = blockIdx.x * 256 + t;
  int b = gid / (H1 * W1); int r = gid % (H1 * W1);
  int py = r / W1; int px = r % W1;
  const float* xb = x + (size_t)b * H0 * W0;
  float patch[4][4];
  int y0 = py * 2 - 1, x0 = px * 2 - 1;
#pragma unroll
  for (int i = 0; i < 4; i++)
#pragma unroll
    for (int j = 0; j < 4; j++) {
      int yy = y0 + i, xx = x0 + j;
      patch[i][j] = (yy >= 0 && yy < H0 && xx >= 0 && xx < W0) ? xb[yy * W0 + xx] : 0.f;
    }
  bf16_t* outp = pooled1 + ((size_t)(b * H1 + py) * W1 + px) * C1;
  for (int c = 0; c < C1; c++) {
    float mx = 0.f;
#pragma unroll
    for (int dy = 0; dy < 2; dy++)
#pragma unroll
      for (int dx = 0; dx < 2; dx++) {
        float a = sb[c];
#pragma unroll
        for (int kh = 0; kh < 3; kh++)
#pragma unroll
          for (int kw = 0; kw < 3; kw++)
            a = fmaf(sw[c * 9 + kh * 3 + kw], patch[kh + dy][kw + dx], a);
        a = (a - sm[c]) * sr[c] * sg[c] + sbe[c];
        a = fmaxf(a, 0.f);
        mx = fmaxf(mx, a);
      }
    outp[c] = (bf16_t)mx;
  }
}

/* ---- pack conv2 weights into WMMA A-fragment lane layout, bf16 ----
   K ordering: k = (kh*3+kw)*24 + ci  (216 valid, padded to 224)         */
__global__ void k_w2pack(const float* __restrict__ w2, bf16_t* __restrict__ wp) {
  int lane = threadIdx.x;       /* 32 */
  int kc = blockIdx.x;          /* 7  */
  int mt = blockIdx.y;          /* 3  */
  int m = lane & 15;
  int co = mt * 16 + m;
  bool hi = lane >= 16;
  for (int e = 0; e < 16; e++) {
    int koff;
    if (!hi) koff = e;
    else     koff = 16 + e;
    int k = kc * 32 + koff;
    float v = 0.f;
    if (k < 216) {
      int tap = k / 24, ci = k % 24;
      int kh = tap / 3, kw = tap % 3;
      v = w2[((co * C1 + ci) * 3 + kh) * 3 + kw];
    }
    wp[((size_t)(mt * NCHUNK + kc) * 32 + lane) * 16 + e] = (bf16_t)v;
  }
}

/* ---- conv2 as im2col GEMM via v_wmma_f32_16x16x32_bf16; one wave per 16x16 tile.
   B fragment assembled as 8 packed dwords: each dword is ONE aligned b32 load
   from the NHWC pooled1 buffer (K pairs never straddle a tap), zeroed by
   v_cndmask for halo/padding — branch-free, register-resident, EXEC stays full. */
__global__ void k_conv2_wmma(const bf16_t* __restrict__ pooled1, const bf16_t* __restrict__ wp,
                             const float* __restrict__ c2b,
                             bf16_t* __restrict__ raw2, float* __restrict__ F) {
  int lane = threadIdx.x;
  int nt = blockIdx.x;          /* 50176 spatial tiles */
  int mt = blockIdx.y;          /* 3 channel tiles     */
  int nidx = lane & 15;
  bool hi = lane >= 16;
  int n = nt * 16 + nidx;
  int b = n / (H1 * W1); int r = n % (H1 * W1);
  int y = r / W1; int x = r % W1;
  size_t imgbase = (size_t)b * H1 * W1;

  v8f acc = {0.f, 0.f, 0.f, 0.f, 0.f, 0.f, 0.f, 0.f};
  const v16bf* wpv = (const v16bf*)(wp + (size_t)(mt * NCHUNK) * 32 * 16);
#pragma unroll
  for (int kc = 0; kc < NCHUNK; kc++) {
    v16bf A = wpv[kc * 32 + lane];
    v8u bu;
#pragma unroll
    for (int j = 0; j < 8; j++) {
      /* compile-time lo/hi variants; runtime select on `hi` is one cndmask each */
      const int kL = kc * 32 + 2 * j;
      const int kH = kL + 16;
      const int tapL = kL / 24, ciL = kL % 24, dyL = tapL / 3 - 1, dxL = tapL % 3 - 1;
      const int tapH = kH / 24, ciH = kH % 24, dyH = tapH / 3 - 1, dxH = tapH % 3 - 1;
      const bool okL = kL < 216, okH = kH < 216;
      int dy = hi ? dyH : dyL;
      int dx = hi ? dxH : dxL;
      int ci = hi ? ciH : ciL;
      bool kok = hi ? okH : okL;
      int yy = y + dy, xx = x + dx;
      bool ok = kok & (yy >= 0) & (yy < H1) & (xx >= 0) & (xx < W1);
      int yc = min(max(yy, 0), H1 - 1);
      int xc = min(max(xx, 0), W1 - 1);
      unsigned int u = *(const unsigned int*)(pooled1 + ((imgbase + (size_t)yc * W1 + xc) * C1 + ci));
      bu[j] = ok ? u : 0u;
    }
    v16bf Bv = __builtin_bit_cast(v16bf, bu);
    acc = __builtin_amdgcn_wmma_f32_16x16x32_bf16(false, A, false, Bv, (short)0, acc,
                                                  false, false);
  }
#pragma unroll
  for (int vv = 0; vv < 8; vv++) {
    int m = vv + (hi ? 8 : 0);
    int co = mt * 16 + m;
    float val = acc[vv] + c2b[co];
    raw2[(((size_t)b * C2 + co) * H1 + y) * W1 + x] = (bf16_t)val;
    float s = val, q = val * val;
#pragma unroll
    for (int msk = 1; msk < 16; msk <<= 1) {
      s += __shfl_xor(s, msk, 32);
      q += __shfl_xor(q, msk, 32);
    }
    if (nidx == 0) {
      atomicAdd(&F[OFF_SUM2 + co], s);
      atomicAdd(&F[OFF_SQ2 + co], q);
    }
  }
}

/* ---- BN2 + ReLU + maxpool2, write x1 bf16 in (b)(c,h,w) GEMM-row layout ---- */
__global__ void k_bn2_pool(const bf16_t* __restrict__ raw2, const float* __restrict__ g,
                           const float* __restrict__ beta, const float* __restrict__ F,
                           bf16_t* __restrict__ x1) {
  int gid = blockIdx.x * 256 + threadIdx.x;
  int b = gid / (C2 * H2 * W2); int r = gid % (C2 * H2 * W2);
  int co = r / (H2 * W2); int rr = r % (H2 * W2);
  int py = rr / W2; int px = rr % W2;
  float m = F[OFF_MEAN2 + co], rs = F[OFF_RS2 + co];
  float sc = rs * g[co], sh = beta[co] - m * sc;
  const bf16_t* rb = raw2 + (((size_t)b * C2 + co) * H1 + py * 2) * W1 + px * 2;
  float a0 = fmaxf(fmaf((float)rb[0],      sc, sh), 0.f);
  float a1 = fmaxf(fmaf((float)rb[1],      sc, sh), 0.f);
  float a2 = fmaxf(fmaf((float)rb[W1],     sc, sh), 0.f);
  float a3 = fmaxf(fmaf((float)rb[W1 + 1], sc, sh), 0.f);
  float mx = fmaxf(fmaxf(a0, a1), fmaxf(a2, a3));
  x1[(size_t)b * KF + (size_t)co * (H2 * W2) + py * W2 + px] = (bf16_t)mx;
}

/* ---- feat = X[64,150528] * W^T[150528,48] via WMMA bf16, K split 96-way ----
   B fragment: 16 consecutive fp32 -> manual RNE bf16 pack into 8 dwords. */
__global__ void k_fconv_wmma(const bf16_t* __restrict__ x1, const float* __restrict__ fw,
                             float* __restrict__ F) {
  int lane = threadIdx.x;
  int kz = blockIdx.x;          /* 96 K-slices of 49 chunks each (4704 total) */
  int nt = blockIdx.y;          /* 3  */
  int mt = blockIdx.z;          /* 4  */
  int nidx = lane & 15;
  bool hi = lane >= 16;
  int bb = mt * 16 + nidx;      /* A-matrix row (batch) */
  int o  = nt * 16 + nidx;      /* B-matrix column (out channel) */
  v8f acc = {0.f, 0.f, 0.f, 0.f, 0.f, 0.f, 0.f, 0.f};
  int kc0 = kz * 49, kc1 = kc0 + 49;
  for (int kc = kc0; kc < kc1; kc++) {
    size_t base = (size_t)bb * KF + (size_t)kc * 32;
    v8bf lo  = *(const v8bf*)(x1 + base + (hi ? 16 : 0));
    v8bf hi8 = *(const v8bf*)(x1 + base + (hi ? 24 : 8));
    v16bf A = __builtin_shufflevector(lo, hi8, 0, 1, 2, 3, 4, 5, 6, 7,
                                      8, 9, 10, 11, 12, 13, 14, 15);
    const float* wb = fw + (size_t)o * KF + (size_t)kc * 32 + (hi ? 16 : 0);
    v8u bu;
#pragma unroll
    for (int j = 0; j < 8; j++) {
      unsigned int l = f32_to_bf16_rne(wb[2 * j]);
      unsigned int h = f32_to_bf16_rne(wb[2 * j + 1]);
      bu[j] = l | (h << 16);
    }
    v16bf Bv = __builtin_bit_cast(v16bf, bu);
    acc = __builtin_amdgcn_wmma_f32_16x16x32_bf16(false, A, false, Bv, (short)0, acc,
                                                  false, false);
  }
#pragma unroll
  for (int vv = 0; vv < 8; vv++) {
    int mm = vv + (hi ? 8 : 0);
    atomicAdd(&F[OFF_FEAT + (mt * 16 + mm) * C2 + o], acc[vv]);
  }
}

/* ---- tiny linear head: feat -> theta params; also emit tx/ty/scale/angle ---- */
__global__ void k_head(const float* __restrict__ fcb, const float* __restrict__ l1w,
                       const float* __restrict__ l1b, const float* __restrict__ l2w,
                       const float* __restrict__ l2b, float* __restrict__ F,
                       float* __restrict__ out_tail) {
  int b = threadIdx.x;
  if (b >= Bn) return;
  float fb[C2];
  for (int j = 0; j < C2; j++) fb[j] = F[OFF_FEAT + b * C2 + j] + fcb[j];
  float h[24];
  for (int i = 0; i < 24; i++) {
    float s = l1b[i];
    for (int j = 0; j < C2; j++) s = fmaf(l1w[i * C2 + j], fb[j], s);
    h[i] = fmaxf(s, 0.f);
  }
  float t4[4];
  for (int k = 0; k < 4; k++) {
    float s = l2b[k];
    for (int i = 0; i < 24; i++) s = fmaf(l2w[k * 24 + i], h[i], s);
    t4[k] = s;
  }
  float tx = t4[0], ty = t4[1], sc = t4[2], ang = t4[3];
  float c = cosf(ang), s = sinf(ang);
  float* th = F + OFF_THETA + b * 6;
  th[0] = sc * c;  th[1] = -sc * s; th[2] = tx;
  th[3] = sc * s;  th[4] = sc * c;  th[5] = ty;
  out_tail[b] = tx; out_tail[64 + b] = ty; out_tail[128 + b] = sc; out_tail[192 + b] = ang;
}

/* ---- affine grid + bilinear grid_sample of the original input ---- */
__global__ void k_sample(const float* __restrict__ x, const float* __restrict__ F,
                         float* __restrict__ out) {
  int gid = blockIdx.x * 256 + threadIdx.x;
  int b = gid / (H0 * W0); int r = gid % (H0 * W0);
  int y = r / W0; int xx = r % W0;
  const float* th = F + OFF_THETA + b * 6;
  float gx = ((2.f * xx + 1.f) / (float)W0) - 1.f;
  float gy = ((2.f * y + 1.f) / (float)H0) - 1.f;
  float gxo = th[0] * gx + th[1] * gy + th[2];
  float gyo = th[3] * gx + th[4] * gy + th[5];
  float ix = ((gxo + 1.f) * (float)W0 - 1.f) * 0.5f;
  float iy = ((gyo + 1.f) * (float)H0 - 1.f) * 0.5f;
  float x0 = floorf(ix), y0 = floorf(iy);
  float wx1 = ix - x0, wx0 = 1.f - wx1;
  float wy1 = iy - y0, wy0 = 1.f - wy1;
  const float* xb = x + (size_t)b * H0 * W0;
  auto tap = [&](float xf, float yf, float wgt) -> float {
    bool valid = (xf >= 0.f) && (xf < (float)W0) && (yf >= 0.f) && (yf < (float)H0);
    int xi = (int)fminf(fmaxf(xf, 0.f), (float)(W0 - 1));
    int yi = (int)fminf(fmaxf(yf, 0.f), (float)(H0 - 1));
    return xb[yi * W0 + xi] * (valid ? wgt : 0.f);
  };
  out[gid] = tap(x0, y0, wx0 * wy0) + tap(x0 + 1.f, y0, wx1 * wy0) +
             tap(x0, y0 + 1.f, wx0 * wy1) + tap(x0 + 1.f, y0 + 1.f, wx1 * wy1);
}

extern "C" void kernel_launch(void* const* d_in, const int* in_sizes, int n_in,
                              void* d_out, int out_size, void* d_ws, size_t ws_size,
                              hipStream_t stream) {
  (void)in_sizes; (void)n_in; (void)out_size; (void)ws_size;
  const float* x   = (const float*)d_in[0];
  const float* c1w = (const float*)d_in[1];
  const float* c1b = (const float*)d_in[2];
  const float* g1  = (const float*)d_in[3];
  const float* b1  = (const float*)d_in[4];
  const float* c2w = (const float*)d_in[5];
  const float* c2b = (const float*)d_in[6];
  const float* g2  = (const float*)d_in[7];
  const float* b2  = (const float*)d_in[8];
  const float* fw  = (const float*)d_in[9];
  const float* fcb = (const float*)d_in[10];
  const float* l1w = (const float*)d_in[11];
  const float* l1b = (const float*)d_in[12];
  const float* l2w = (const float*)d_in[13];
  const float* l2b = (const float*)d_in[14];
  float* out = (float*)d_out;
  char* ws = (char*)d_ws;
  float*  F       = (float*)ws;
  bf16_t* wp      = (bf16_t*)(ws + WPACK_BYTE);
  bf16_t* pooled1 = (bf16_t*)(ws + POOLED1_BYTE);
  bf16_t* raw2    = (bf16_t*)(ws + CONV2RAW_BYTE);
  bf16_t* x1      = (bf16_t*)(ws + X1_BYTE);

  k_zero<<<(ZERO_COUNT + 255) / 256, 256, 0, stream>>>(F, ZERO_COUNT);
  k_conv1_stats<<<12544, 256, 0, stream>>>(x, c1w, c1b, F);
  k_finalize<<<1, 64, 0, stream>>>(F, OFF_SUM1, OFF_SQ1, OFF_MEAN1, OFF_RS1, C1,
                                   1.f / (float)((size_t)Bn * H0 * W0));
  k_conv1_apply<<<3136, 256, 0, stream>>>(x, c1w, c1b, g1, b1, F, pooled1);
  k_w2pack<<<dim3(NCHUNK, 3), 32, 0, stream>>>(c2w, wp);
  k_conv2_wmma<<<dim3(50176, 3), 32, 0, stream>>>(pooled1, wp, c2b, raw2, F);
  k_finalize<<<1, 64, 0, stream>>>(F, OFF_SUM2, OFF_SQ2, OFF_MEAN2, OFF_RS2, C2,
                                   1.f / (float)((size_t)Bn * H1 * W1));
  k_bn2_pool<<<37632, 256, 0, stream>>>(raw2, g2, b2, F, x1);
  k_fconv_wmma<<<dim3(96, 3, 4), 32, 0, stream>>>(x1, fw, F);
  k_head<<<1, 64, 0, stream>>>(fcb, l1w, l1b, l2w, l2b, F, out + (size_t)Bn * H0 * W0);
  k_sample<<<12544, 256, 0, stream>>>(x, F, out);
}
